// Model_52321291599987
// MI455X (gfx1250) — matrix operand order repaired and checked by reading, not compiled, not run on hardware
//
#include <hip/hip_runtime.h>

typedef __attribute__((ext_vector_type(16))) __bf16 v16bf;
typedef __attribute__((ext_vector_type(8)))  __bf16 v8bf;
typedef __attribute__((ext_vector_type(8)))  float  v8f;

#define NTOK 676
#define KSEL 540
#define KLOW 136
#define CDIM 768
#define HN   12
#define DH   64
#define LL   17
#define BN   32
#define NSEQ 677
#define SSEQ 542
#define SPAD 544
#define C3   2304
#define C4   3072

// ---------------- small helpers ----------------

static __device__ __forceinline__ v16bf bf_load16(const __bf16* p) {
  v8bf lo = *(const v8bf*)p;
  v8bf hi = *(const v8bf*)(p + 8);
  return __builtin_shufflevector(lo, hi, 0,1,2,3,4,5,6,7,8,9,10,11,12,13,14,15);
}
static __device__ __forceinline__ v16bf bf_load8x2(const __bf16* p0, const __bf16* p1) {
  v8bf lo = *(const v8bf*)p0;
  v8bf hi = *(const v8bf*)p1;
  return __builtin_shufflevector(lo, hi, 0,1,2,3,4,5,6,7,8,9,10,11,12,13,14,15);
}
static __device__ __forceinline__ v8f zero8f() {
  v8f r; for (int i = 0; i < 8; ++i) r[i] = 0.f; return r;
}
static __device__ __forceinline__ v16bf zero16bf() {
  v16bf r; for (int i = 0; i < 16; ++i) r[i] = (__bf16)0.f; return r;
}
static __device__ __forceinline__ v8f wmma_bf16(v16bf a, v16bf b, v8f c) {
  return __builtin_amdgcn_wmma_f32_16x16x32_bf16(false, a, false, b, (short)0, c, false, false);
}

// CDNA5 async global->LDS copy (ASYNCcnt-tracked DMA into LDS), 16 bytes per lane.
static __device__ __forceinline__ void async_cp16(void* lds_dst, const void* gsrc) {
  unsigned lds = (unsigned)(unsigned long long)lds_dst;  // low 32 bits = LDS byte address
  asm volatile("global_load_async_to_lds_b128 %0, %1, off"
               :: "v"(lds), "v"(gsrc)
               : "memory");
}
static __device__ __forceinline__ void wait_async0() {
  asm volatile("s_wait_asynccnt 0" ::: "memory");
}

// ---------------- preprocessing kernels ----------------

__global__ void k_langmean(const float* __restrict__ lang, float* __restrict__ lm) {
  int i = blockIdx.x * 256 + threadIdx.x;
  if (i >= BN * CDIM) return;
  float s = 0.f;
  for (int l = 0; l < LL; ++l) s += lang[(size_t)l * BN * CDIM + i];
  lm[i] = s * (1.0f / LL);
}

// score[n*B+b] = dot(x[1+n][b][:], langmean[b][:])  (mean over L folded into langmean)
__global__ void k_score(const float* __restrict__ x, const float* __restrict__ lm,
                        float* __restrict__ score) {
  int gid = blockIdx.x * 8 + (threadIdx.x >> 5);
  if (gid >= NTOK * BN) return;
  int lane = threadIdx.x & 31;
  int n = gid / BN, b = gid - n * BN;
  const float* xr = x + ((size_t)(1 + n) * BN + b) * CDIM;
  const float* lr = lm + (size_t)b * CDIM;
  float acc = 0.f;
  for (int c = lane; c < CDIM; c += 32) acc += xr[c] * lr[c];
  for (int off = 16; off > 0; off >>= 1) acc += __shfl_down(acc, off);
  if (lane == 0) score[gid] = acc;
}

// Per-batch exact top-K via bitonic sort (desc score, asc index on ties == stable argsort(-score))
__global__ void k_topk(const float* __restrict__ score, int* __restrict__ flags) {
  __shared__ float sv[1024];
  __shared__ int   si[1024];
  int b = blockIdx.x;
  for (int i = threadIdx.x; i < 1024; i += 256) {
    sv[i] = (i < NTOK) ? score[i * BN + b] : -3.0e38f;
    si[i] = i;
  }
  __syncthreads();
  for (int k = 2; k <= 1024; k <<= 1) {
    for (int j = k >> 1; j > 0; j >>= 1) {
      for (int i = threadIdx.x; i < 1024; i += 256) {
        int ixj = i ^ j;
        if (ixj > i) {
          bool dirFirst = ((i & k) == 0);
          float va = sv[i], vb = sv[ixj];
          int ia = si[i], ib = si[ixj];
          bool inOrder = (va > vb) || (va == vb && ia < ib);
          if (inOrder != dirFirst) {
            sv[i] = vb; sv[ixj] = va; si[i] = ib; si[ixj] = ia;
          }
        }
      }
      __syncthreads();
    }
  }
  for (int i = threadIdx.x; i < NTOK; i += 256)
    flags[si[i] * BN + b] = (i < KSEL) ? 1 : 0;
}

// Ordered stream compaction (single block): idx_hi, idx_lo in ascending flat order,
// plus inverse map src_row[flat] -> attn_out row index.
__global__ void k_compact(const int* __restrict__ flags, int* __restrict__ idx_hi,
                          int* __restrict__ idx_lo, int* __restrict__ src_row) {
  __shared__ int sscan[256];
  __shared__ int base_hi, base_lo;
  if (threadIdx.x == 0) { base_hi = 0; base_lo = 0; }
  __syncthreads();
  const int total = NTOK * BN;
  for (int start = 0; start < total; start += 256) {
    int idx = start + (int)threadIdx.x;
    int f = (idx < total) ? flags[idx] : 0;
    sscan[threadIdx.x] = f;
    __syncthreads();
    for (int off = 1; off < 256; off <<= 1) {
      int v = sscan[threadIdx.x];
      if ((int)threadIdx.x >= off) v += sscan[threadIdx.x - off];
      __syncthreads();
      sscan[threadIdx.x] = v;
      __syncthreads();
    }
    int excl_hi = sscan[threadIdx.x] - f;
    int chunk_hi = sscan[255];
    int cnt = total - start; if (cnt > 256) cnt = 256;
    if (idx < total) {
      if (f) {
        int t = base_hi + excl_hi;
        idx_hi[t] = idx;
        src_row[idx] = BN + t;                       // attn rows [1..K] block
      } else {
        int tl = base_lo + ((int)threadIdx.x - excl_hi);
        idx_lo[tl] = idx;
        src_row[idx] = (SSEQ - 1) * BN + (tl / KLOW); // repeat of attn_out[-1]
      }
    }
    __syncthreads();
    if (threadIdx.x == 0) { base_hi += chunk_hi; base_lo += (cnt - chunk_hi); }
    __syncthreads();
  }
}

// low_pool for reshaped column j: softmax over KLOW scores, weighted sum of rows
__global__ void k_lowpool(const float* __restrict__ x, const float* __restrict__ score,
                          const int* __restrict__ idx_lo, float* __restrict__ lowpool) {
  __shared__ float wgt[KLOW];
  __shared__ int   fl[KLOW];
  int j = blockIdx.x;
  for (int i = threadIdx.x; i < KLOW; i += 256) {
    int flat = idx_lo[i * BN + j];
    fl[i] = flat;
    wgt[i] = score[flat];
  }
  __syncthreads();
  if (threadIdx.x == 0) {
    float mx = -3.0e38f;
    for (int i = 0; i < KLOW; ++i) mx = fmaxf(mx, wgt[i]);
    float s = 0.f;
    for (int i = 0; i < KLOW; ++i) { float e = __expf(wgt[i] - mx); wgt[i] = e; s += e; }
    float inv = 1.f / s;
    for (int i = 0; i < KLOW; ++i) wgt[i] *= inv;
  }
  __syncthreads();
  for (int c = threadIdx.x; c < CDIM; c += 256) {
    float acc = 0.f;
    for (int i = 0; i < KLOW; ++i) {
      int flat = fl[i];
      int n = flat / BN, b = flat - n * BN;
      acc += wgt[i] * x[((size_t)(1 + n) * BN + b) * CDIM + c];
    }
    lowpool[(size_t)j * CDIM + c] = acc;
  }
}

// Gather dx row (cls / selected token / low_pool), LayerNorm, emit bf16
__global__ void k_dxln(const float* __restrict__ x, const float* __restrict__ lowpool,
                       const int* __restrict__ idx_hi, const float* __restrict__ g,
                       const float* __restrict__ be, __bf16* __restrict__ dxln) {
  __shared__ float red[256];
  __shared__ float smean, srstd;
  int row = blockIdx.x;                   // row = p*B + b
  int p = row / BN, b = row - p * BN;
  const float* src;
  if (p == 0) src = x + (size_t)b * CDIM;
  else if (p == SSEQ - 1) src = lowpool + (size_t)b * CDIM;
  else {
    int flat = idx_hi[(p - 1) * BN + b];
    int n = flat / BN, b2 = flat - n * BN;
    src = x + ((size_t)(1 + n) * BN + b2) * CDIM;
  }
  float s = 0.f;
  for (int c = threadIdx.x; c < CDIM; c += 256) s += src[c];
  red[threadIdx.x] = s; __syncthreads();
  for (int off = 128; off > 0; off >>= 1) {
    if ((int)threadIdx.x < off) red[threadIdx.x] += red[threadIdx.x + off];
    __syncthreads();
  }
  if (threadIdx.x == 0) smean = red[0] * (1.0f / CDIM);
  __syncthreads();
  float m = smean;
  float v = 0.f;
  for (int c = threadIdx.x; c < CDIM; c += 256) { float d = src[c] - m; v += d * d; }
  red[threadIdx.x] = v; __syncthreads();
  for (int off = 128; off > 0; off >>= 1) {
    if ((int)threadIdx.x < off) red[threadIdx.x] += red[threadIdx.x + off];
    __syncthreads();
  }
  if (threadIdx.x == 0) srstd = rsqrtf(red[0] * (1.0f / CDIM) + 1e-5f);
  __syncthreads();
  float rs = srstd;
  __bf16* dst = dxln + (size_t)row * CDIM;
  for (int c = threadIdx.x; c < CDIM; c += 256)
    dst[c] = (__bf16)((src[c] - m) * rs * g[c] + be[c]);
}

// Plain row LayerNorm (fp32 in, bf16 out)
__global__ void k_ln_rows(const float* __restrict__ in, const float* __restrict__ g,
                          const float* __restrict__ be, __bf16* __restrict__ outb) {
  __shared__ float red[256];
  __shared__ float smean, srstd;
  int row = blockIdx.x;
  const float* src = in + (size_t)row * CDIM;
  float s = 0.f;
  for (int c = threadIdx.x; c < CDIM; c += 256) s += src[c];
  red[threadIdx.x] = s; __syncthreads();
  for (int off = 128; off > 0; off >>= 1) {
    if ((int)threadIdx.x < off) red[threadIdx.x] += red[threadIdx.x + off];
    __syncthreads();
  }
  if (threadIdx.x == 0) smean = red[0] * (1.0f / CDIM);
  __syncthreads();
  float m = smean;
  float v = 0.f;
  for (int c = threadIdx.x; c < CDIM; c += 256) { float d = src[c] - m; v += d * d; }
  red[threadIdx.x] = v; __syncthreads();
  for (int off = 128; off > 0; off >>= 1) {
    if ((int)threadIdx.x < off) red[threadIdx.x] += red[threadIdx.x + off];
    __syncthreads();
  }
  if (threadIdx.x == 0) srstd = rsqrtf(red[0] * (1.0f / CDIM) + 1e-5f);
  __syncthreads();
  float rs = srstd;
  __bf16* dst = outb + (size_t)row * CDIM;
  for (int c = threadIdx.x; c < CDIM; c += 256)
    dst[c] = (__bf16)((src[c] - m) * rs * g[c] + be[c]);
}

__global__ void k_cvt(const float* __restrict__ in, __bf16* __restrict__ outb, size_t n) {
  size_t i = (size_t)blockIdx.x * 256 + threadIdx.x;
  if (i < n) outb[i] = (__bf16)in[i];
}

__global__ void k_padzero(__bf16* qb, __bf16* kb, __bf16* vtb) {
  int i = blockIdx.x * 256 + threadIdx.x;
  const int tot = BN * HN * DH * 2;
  if (i >= tot) return;
  int d = i & 63; int rest = i >> 6;
  int t = SSEQ + (rest & 1); int bh = rest >> 1;
  qb[((size_t)bh * SPAD + t) * DH + d] = (__bf16)0.f;
  kb[((size_t)bh * SPAD + t) * DH + d] = (__bf16)0.f;
  vtb[((size_t)bh * DH + d) * SPAD + t] = (__bf16)0.f;
}

// ---------------- WMMA GEMM: C(MxN) = A(MxK) * W(NxK)^T + bias ----------------
// LDS-tiled 64x128x32, double-buffered via CDNA5 async global->LDS DMA (ASYNCcnt).
// 8 waves per block; each wave owns a 32x32 accumulator patch (2x2 WMMA tiles).
// mode 0: fp32 out   mode 1: bf16 out   mode 2: bf16 out with h*sigmoid(1.702h)
// mode 3: scatter qkv into q/k/v^T layouts   mode 4: fp32 accumulate (+=)
#define ASTR 40   // padded LDS row stride (elements) -> conflict-free fragment reads
__global__ void __launch_bounds__(256)
k_gemm(const __bf16* __restrict__ A, const __bf16* __restrict__ W,
       const float* __restrict__ bias, int M, int N, int Kd, int mode,
       float* outf, __bf16* outb, __bf16* qb, __bf16* kb, __bf16* vtb) {
  __shared__ __bf16 a_s[2][64][ASTR];
  __shared__ __bf16 b_s[2][128][ASTR];

  int tilesN = N >> 7;                         // N / 128
  int tm0 = (int)blockIdx.x / tilesN * 64;     // block M origin
  int tn0 = ((int)blockIdx.x % tilesN) * 128;  // block N origin
  int tid = threadIdx.x;
  int wave = tid >> 5, lane = tid & 31;
  int nloc = lane & 15, hf = lane >> 4;
  int wm = (wave & 1) * 32;                    // wave's M sub-origin within block
  int wn = (wave >> 1) * 32;                   // wave's N sub-origin within block

  // loader assignments (16B async chunks)
  int arow = tid >> 2, ac8 = (tid & 3) * 8;    // A: 64 rows x 4 chunks
  int arowg = tm0 + arow; if (arowg >= M) arowg = M - 1;
  const __bf16* aldg = A + (size_t)arowg * Kd + ac8;
  int bc0 = tid, bc1 = tid + 256;              // B: 128 rows x 4 chunks = 512
  int brow0 = bc0 >> 2, bc80 = (bc0 & 3) * 8;
  int brow1 = bc1 >> 2, bc81 = (bc1 & 3) * 8;
  const __bf16* bldg0 = W + (size_t)(tn0 + brow0) * Kd + bc80;
  const __bf16* bldg1 = W + (size_t)(tn0 + brow1) * Kd + bc81;

  auto stage = [&](int buf, int kk) {
    async_cp16(&a_s[buf][arow][ac8], aldg + kk);
    async_cp16(&b_s[buf][brow0][bc80], bldg0 + kk);
    async_cp16(&b_s[buf][brow1][bc81], bldg1 + kk);
  };

  v8f acc[2][2];
  for (int i = 0; i < 2; ++i)
    for (int j = 0; j < 2; ++j) acc[i][j] = zero8f();

  int nk = Kd >> 5;
  stage(0, 0);
  wait_async0();
  __syncthreads();
  for (int ks = 0; ks < nk; ++ks) {
    int cur = ks & 1;
    if (ks + 1 < nk) stage(cur ^ 1, (ks + 1) << 5);
    // fragments from LDS
    v16bf af[2], bf[2];
    for (int i = 0; i < 2; ++i) {
      const __bf16* pa = &a_s[cur][wm + i * 16 + nloc][hf * 8];
      af[i] = bf_load8x2(pa, pa + 16);
    }
    for (int j = 0; j < 2; ++j) {
      const __bf16* pb = &b_s[cur][wn + j * 16 + nloc][hf * 8];
      bf[j] = bf_load8x2(pb, pb + 16);
    }
    for (int i = 0; i < 2; ++i)
      for (int j = 0; j < 2; ++j)
        acc[i][j] = wmma_bf16(af[i], bf[j], acc[i][j]);
    wait_async0();
    __syncthreads();
  }

  for (int i = 0; i < 2; ++i)
    for (int j = 0; j < 2; ++j) {
      int n = tn0 + wn + j * 16 + nloc;
      float bv = bias ? bias[n] : 0.f;
      for (int r = 0; r < 8; ++r) {
        int m = tm0 + wm + i * 16 + r + hf * 8;
        if (m >= M) continue;
        float v = acc[i][j][r] + bv;
        if (mode == 0) {
          outf[(size_t)m * N + n] = v;
        } else if (mode == 1) {
          outb[(size_t)m * N + n] = (__bf16)v;
        } else if (mode == 2) {
          float gsig = v / (1.f + __expf(-1.702f * v));
          outb[(size_t)m * N + n] = (__bf16)gsig;
        } else if (mode == 4) {
          outf[(size_t)m * N + n] += v;
        } else { // mode 3: scatter into q/k/v^T bf16 layouts
          int which = n / CDIM;
          int cc = n - which * CDIM;
          int h = cc >> 6, d = cc & 63;
          int p = m / BN, b = m - (m / BN) * BN;
          if (which == 0)      qb[(((size_t)(b * HN + h)) * SPAD + p) * DH + d] = (__bf16)v;
          else if (which == 1) kb[(((size_t)(b * HN + h)) * SPAD + p) * DH + d] = (__bf16)v;
          else                 vtb[(((size_t)(b * HN + h)) * DH + d) * SPAD + p] = (__bf16)v;
        }
      }
    }
}

// ---------------- flash attention (per b,h block; 8 waves; WMMA bf16) ----------------
__global__ void __launch_bounds__(256)
k_attn(const __bf16* __restrict__ qbuf, const __bf16* __restrict__ kbuf,
       const __bf16* __restrict__ vtbuf, __bf16* __restrict__ obuf) {
  __shared__ float  sc[8][512];      // 16x32 score chunk per wave
  __shared__ __bf16 pbv[8][512];     // 16x32 bf16 p chunk (row-major)
  __shared__ float  mrow[8][16], srow[8][16], frow[8][16];
  int bh = blockIdx.x;
  int b = bh / HN, h = bh - b * HN;
  int wave = threadIdx.x >> 5, lane = threadIdx.x & 31;
  int nloc = lane & 15, hf = lane >> 4;
  const __bf16* qp = qbuf + (size_t)bh * SPAD * DH;
  const __bf16* kp = kbuf + (size_t)bh * SPAD * DH;
  const __bf16* vp = vtbuf + (size_t)bh * DH * SPAD;
  for (int qi = 0; qi < 5; ++qi) {
    int qt = qi * 8 + wave;
    bool active = (qt < 34);
    v16bf aq0 = zero16bf(), aq1 = zero16bf();
    v8f oacc[4];
    for (int ct = 0; ct < 4; ++ct) oacc[ct] = zero8f();
    if (active) {
      const __bf16* qr = qp + (size_t)(qt * 16 + nloc) * DH + hf * 8;
      aq0 = bf_load8x2(qr, qr + 16);
      aq1 = bf_load8x2(qr + 32, qr + 48);
      if (lane < 16) { mrow[wave][lane] = -3.0e38f; srow[wave][lane] = 0.f; }
    }
    __syncthreads();
    for (int kt = 0; kt < 17; ++kt) {
      int key0 = kt * 32;
      if (active) {
        for (int t = 0; t < 2; ++t) {
          v8f acc = zero8f();
          const __bf16* kr = kp + (size_t)(key0 + t * 16 + nloc) * DH + hf * 8;
          v16bf b0 = bf_load8x2(kr, kr + 16);
          v16bf b1 = bf_load8x2(kr + 32, kr + 48);
          acc = wmma_bf16(aq0, b0, acc);
          acc = wmma_bf16(aq1, b1, acc);
          for (int r = 0; r < 8; ++r)
            sc[wave][(r + hf * 8) * 32 + t * 16 + nloc] = acc[r] * 0.125f;
        }
      }
      __syncthreads();
      if (active && lane < 16) {
        float* rowp = &sc[wave][lane * 32];
        float tm = -3.0e38f;
        for (int jj = 0; jj < 32; ++jj) {
          float v = rowp[jj];
          if (key0 + jj >= SSEQ) v = -3.0e38f;
          rowp[jj] = v;
          tm = fmaxf(tm, v);
        }
        float mo = mrow[wave][lane];
        float mn = fmaxf(mo, tm);
        float corr = __expf(mo - mn);
        float ts = 0.f;
        for (int jj = 0; jj < 32; ++jj) {
          float pv = __expf(rowp[jj] - mn);
          ts += pv;
          pbv[wave][lane * 32 + jj] = (__bf16)pv;
        }
        srow[wave][lane] = srow[wave][lane] * corr + ts;
        mrow[wave][lane] = mn;
        frow[wave][lane] = corr;
      }
      __syncthreads();
      if (active) {
        float fr[8];
        for (int r = 0; r < 8; ++r) fr[r] = frow[wave][r + hf * 8];
        for (int ct = 0; ct < 4; ++ct)
          for (int r = 0; r < 8; ++r) oacc[ct][r] = oacc[ct][r] * fr[r];
        const __bf16* pr = &pbv[wave][nloc * 32 + hf * 8];
        v16bf ap = bf_load8x2(pr, pr + 16);
        for (int ct = 0; ct < 4; ++ct) {
          const __bf16* vr = vp + (size_t)(ct * 16 + nloc) * SPAD + key0 + hf * 8;
          v16bf bv = bf_load8x2(vr, vr + 16);
          oacc[ct] = wmma_bf16(ap, bv, oacc[ct]);
        }
      }
      __syncthreads();
    }
    if (active) {
      float inv[8];
      for (int r = 0; r < 8; ++r) inv[r] = 1.f / srow[wave][r + hf * 8];
      for (int ct = 0; ct < 4; ++ct)
        for (int r = 0; r < 8; ++r) {
          int row = qt * 16 + r + hf * 8;
          if (row < SSEQ)
            obuf[((size_t)row * BN + b) * CDIM + h * DH + ct * 16 + nloc] =
                (__bf16)(oacc[ct][r] * inv[r]);
        }
    }
    __syncthreads();
  }
}

// x2 = x + scattered attention residual
__global__ void k_scatter(const float* __restrict__ x, const float* __restrict__ attno,
                          const int* __restrict__ src_row, float* __restrict__ out) {
  size_t i = (size_t)blockIdx.x * 256 + threadIdx.x;
  const size_t tot = (size_t)NSEQ * BN * CDIM;
  if (i >= tot) return;
  int c = (int)(i % CDIM);
  size_t rb = i / CDIM;
  int b = (int)(rb % BN);
  int p = (int)(rb / BN);
  float add;
  if (p == 0) add = attno[(size_t)b * CDIM + c];
  else        add = attno[(size_t)src_row[(p - 1) * BN + b] * CDIM + c];
  out[i] = x[i] + add;
}

// ---------------- host launcher ----------------

extern "C" void kernel_launch(void* const* d_in, const int* in_sizes, int n_in,
                              void* d_out, int out_size, void* d_ws, size_t ws_size,
                              hipStream_t stream) {
  (void)in_sizes; (void)n_in; (void)out_size; (void)ws_size;
  const float* x      = (const float*)d_in[0];
  const float* lang   = (const float*)d_in[1];
  const float* ln1_g  = (const float*)d_in[2];
  const float* ln1_b  = (const float*)d_in[3];
  const float* w_qkv  = (const float*)d_in[4];
  const float* b_qkv  = (const float*)d_in[5];
  const float* w_out  = (const float*)d_in[6];
  const float* b_out  = (const float*)d_in[7];
  const float* ln2_g  = (const float*)d_in[8];
  const float* ln2_b  = (const float*)d_in[9];
  const float* w_fc   = (const float*)d_in[10];
  const float* b_fc   = (const float*)d_in[11];
  const float* w_proj = (const float*)d_in[12];
  const float* b_proj = (const float*)d_in[13];
  float* out = (float*)d_out;

  char* base = (char*)d_ws;
  size_t off = 0;
  auto alloc = [&](size_t bytes) -> void* {
    void* p = base + off;
    off = (off + bytes + 255) & ~(size_t)255;
    return p;
  };
  float*  langmean = (float*)alloc((size_t)BN * CDIM * 4);
  float*  score    = (float*)alloc((size_t)NTOK * BN * 4);
  int*    flags    = (int*)alloc((size_t)NTOK * BN * 4);
  int*    idx_hi   = (int*)alloc((size_t)KSEL * BN * 4);
  int*    idx_lo   = (int*)alloc((size_t)KLOW * BN * 4);
  int*    src_row  = (int*)alloc((size_t)NTOK * BN * 4);
  float*  lowpool  = (float*)alloc((size_t)BN * CDIM * 4);
  __bf16* wqkv_bf  = (__bf16*)alloc((size_t)C3 * CDIM * 2);
  __bf16* wout_bf  = (__bf16*)alloc((size_t)CDIM * CDIM * 2);
  __bf16* wfc_bf   = (__bf16*)alloc((size_t)C4 * CDIM * 2);
  __bf16* wproj_bf = (__bf16*)alloc((size_t)CDIM * C4 * 2);
  __bf16* dxln     = (__bf16*)alloc((size_t)SSEQ * BN * CDIM * 2);
  __bf16* qb       = (__bf16*)alloc((size_t)BN * HN * SPAD * DH * 2);
  __bf16* kb       = (__bf16*)alloc((size_t)BN * HN * SPAD * DH * 2);
  __bf16* vtb      = (__bf16*)alloc((size_t)BN * HN * DH * SPAD * 2);
  __bf16* obuf     = (__bf16*)alloc((size_t)SSEQ * BN * CDIM * 2);
  float*  attno    = (float*)alloc((size_t)SSEQ * BN * CDIM * 4);
  __bf16* ln2buf   = (__bf16*)alloc((size_t)NSEQ * BN * CDIM * 2);
  __bf16* hact     = (__bf16*)alloc((size_t)NSEQ * BN * C4 * 2);

  const int MB = SSEQ * BN;   // 17344  (divisible by 64)
  const int MN = NSEQ * BN;   // 21664  (64-remainder handled by guards)

  // 1) token scoring + selection
  k_langmean<<<(BN * CDIM + 255) / 256, 256, 0, stream>>>(lang, langmean);
  k_score<<<(NTOK * BN + 7) / 8, 256, 0, stream>>>(x, langmean, score);
  k_topk<<<BN, 256, 0, stream>>>(score, flags);
  k_compact<<<1, 256, 0, stream>>>(flags, idx_hi, idx_lo, src_row);
  k_lowpool<<<BN, 256, 0, stream>>>(x, score, idx_lo, lowpool);
  k_dxln<<<MB, 256, 0, stream>>>(x, lowpool, idx_hi, ln1_g, ln1_b, dxln);

  // 2) weight conversion + pad zero
  k_cvt<<<((size_t)C3 * CDIM + 255) / 256, 256, 0, stream>>>(w_qkv, wqkv_bf, (size_t)C3 * CDIM);
  k_cvt<<<((size_t)CDIM * CDIM + 255) / 256, 256, 0, stream>>>(w_out, wout_bf, (size_t)CDIM * CDIM);
  k_cvt<<<((size_t)C4 * CDIM + 255) / 256, 256, 0, stream>>>(w_fc, wfc_bf, (size_t)C4 * CDIM);
  k_cvt<<<((size_t)CDIM * C4 + 255) / 256, 256, 0, stream>>>(w_proj, wproj_bf, (size_t)CDIM * C4);
  k_padzero<<<(BN * HN * DH * 2 + 255) / 256, 256, 0, stream>>>(qb, kb, vtb);

  // 3) qkv projection (WMMA + async-LDS double buffering), scattered to q/k/v^T layouts
  {
    long blocks = (long)((MB + 63) / 64) * (C3 / 128);
    k_gemm<<<(unsigned)blocks, 256, 0, stream>>>(
        dxln, wqkv_bf, b_qkv, MB, C3, CDIM, 3, nullptr, nullptr, qb, kb, vtb);
  }

  // 4) attention
  k_attn<<<BN * HN, 256, 0, stream>>>(qb, kb, vtb, obuf);

  // 5) output projection (WMMA) -> fp32
  {
    long blocks = (long)((MB + 63) / 64) * (CDIM / 128);
    k_gemm<<<(unsigned)blocks, 256, 0, stream>>>(
        obuf, wout_bf, b_out, MB, CDIM, CDIM, 0, attno, nullptr, nullptr, nullptr, nullptr);
  }

  // 6) scatter residual -> x2 in d_out
  k_scatter<<<(unsigned)(((size_t)NSEQ * BN * CDIM + 255) / 256), 256, 0, stream>>>(
      x, attno, src_row, out);

  // 7) FFN: LN2 -> FC (fused gelu-sigmoid) -> proj accumulate into d_out
  k_ln_rows<<<MN, 256, 0, stream>>>(out, ln2_g, ln2_b, ln2buf);
  {
    long blocks = (long)((MN + 63) / 64) * (C4 / 128);
    k_gemm<<<(unsigned)blocks, 256, 0, stream>>>(
        ln2buf, wfc_bf, b_fc, MN, C4, CDIM, 2, nullptr, hact, nullptr, nullptr, nullptr);
  }
  {
    long blocks = (long)((MN + 63) / 64) * (CDIM / 128);
    k_gemm<<<(unsigned)blocks, 256, 0, stream>>>(
        hact, wproj_bf, b_proj, MN, CDIM, C4, 4, out, nullptr, nullptr, nullptr, nullptr);
  }
}
